// GAT_1451698946828
// MI455X (gfx1250) — matrix operand order repaired and checked by reading, not compiled, not run on hardware
//
#include <hip/hip_runtime.h>

#define NN   4096
#define FIN  128
#define DH   64
#define NH   3
#define NCH  (NN / 32)
#define GALPHA 0.01f
#define NEGC (-1e30f)

typedef __attribute__((ext_vector_type(16))) _Float16 v16h;
typedef __attribute__((ext_vector_type(8)))  _Float16 v8h;
typedef __attribute__((ext_vector_type(8)))  float    v8f;
typedef __attribute__((ext_vector_type(4)))  unsigned int v4u;
typedef __attribute__((ext_vector_type(8)))  int v8i;
typedef __attribute__((ext_vector_type(4)))  int v4i;

__device__ __forceinline__ float warp_sum(float v) {
#pragma unroll
  for (int off = 16; off; off >>= 1) v += __shfl_xor(v, off, 32);
  return v;
}

// TDM: DMA one 16-row x 32-col f32 tile of adj (row stride NN) into LDS.
// D# per ISA 8.3/8.4: group0 = {count=1 | type=2, lds_addr, global_addr},
// group1 = {data_size=4B, tensor_dim0=tensor_dim1=4096, tile_dim0=32,
//           tile_dim1=16, tensor_dim0_stride=4096}.
__device__ __forceinline__ void tdm_load_adj_tile(const float* gptr, unsigned lds_addr) {
  unsigned long long ga = (unsigned long long)(uintptr_t)gptr;
  v4u g0 = { 1u,                                   // count=1, user mode
             lds_addr,                             // LDS byte address
             (unsigned)ga,                         // global_addr[31:0]
             (unsigned)((ga >> 32) & 0x01FFFFFFu) | 0x80000000u }; // addr[56:32] | type=2
  v8i g1 = { 0x20000,                              // data_size = 2 (4 bytes)
             0x10000000,                           // tensor_dim0[15:0]=4096 in [31:16]
             0x10000000,                           // dim0 hi=0, tensor_dim1[15:0]=4096
             32 << 16,                             // dim1 hi=0, tile_dim0=32
             16,                                   // tile_dim1=16, tile_dim2=0
             NN,                                   // tensor_dim0_stride lo = 4096
             0, 0 };
  v4i gz = { 0, 0, 0, 0 };
#if __clang_major__ >= 23
  v8i gz8 = { 0, 0, 0, 0, 0, 0, 0, 0 };
  __builtin_amdgcn_tensor_load_to_lds(g0, g1, gz, gz, gz8, 0);
#else
  __builtin_amdgcn_tensor_load_to_lds(g0, g1, gz, gz, 0);
#endif
}

// ---------------------------------------------------------------------------
// 1) x = feat / rowsum(feat), cast to f16.  One wave per node.
// ---------------------------------------------------------------------------
__global__ void k_rownorm(const float* __restrict__ feat, _Float16* __restrict__ xh) {
  int wave = (blockIdx.x * blockDim.x + threadIdx.x) >> 5;
  int lane = threadIdx.x & 31;
  if (wave >= NN) return;
  const float* row = feat + (size_t)wave * FIN;
  float v[4]; float s = 0.f;
#pragma unroll
  for (int j = 0; j < 4; ++j) { v[j] = row[lane + 32 * j]; s += v[j]; }
  s = warp_sum(s);
  float inv = 1.f / s;
  _Float16* orow = xh + (size_t)wave * FIN;
#pragma unroll
  for (int j = 0; j < 4; ++j) orow[lane + 32 * j] = (_Float16)(v[j] * inv);
}

// ---------------------------------------------------------------------------
// 1b) Swizzle W into WMMA B-fragment layout:
//     wsw[frag=((h*4+kk)*4+dt)][lane][i] = W[h][kk*32 + 16*(lane>>4) + i][dt*16 + (lane&15)]
//     One wave per fragment (48 fragments); each lane stores one contiguous v16h.
// ---------------------------------------------------------------------------
__global__ void k_prepw(const float* __restrict__ Wg, _Float16* __restrict__ wsw) {
  int frag = blockIdx.x * (blockDim.x >> 5) + (threadIdx.x >> 5);
  int lane = threadIdx.x & 31;
  if (frag >= 48) return;
  int dt = frag & 3;
  int kk = (frag >> 2) & 3;
  int h  = frag >> 4;
  int n = lane & 15, hl = lane >> 4;
  v16h b;
#pragma unroll
  for (int i = 0; i < 16; ++i)
    b[i] = (_Float16)Wg[((size_t)h * FIN + kk * 32 + hl * 16 + i) * DH + dt * 16 + n];
  *(v16h*)(wsw + (size_t)frag * 512 + lane * 16) = b;
}

// ---------------------------------------------------------------------------
// 2) Wh[h] = x @ W[h] via v_wmma_f32_16x16x32_f16.  One wave = one 16x16 tile.
//    Writes f32 Wh (for e-dots) and f16 Wh *pre-swizzled into B-fragment
//    layout* for the attention kernel (contiguous v8h store per lane).
// ---------------------------------------------------------------------------
__global__ void k_proj(const _Float16* __restrict__ xh, const _Float16* __restrict__ wsw,
                       float* __restrict__ whf, _Float16* __restrict__ whh) {
  int tile = blockIdx.x * (blockDim.x >> 5) + (threadIdx.x >> 5);
  int lane = threadIdx.x & 31;
  int mt = tile & 255;
  int dt = (tile >> 8) & 3;
  int h  = tile >> 10;
  int m  = lane & 15, hl = lane >> 4;
  const v16h* wv = (const v16h*)wsw;
  v8f c = {};
#pragma unroll
  for (int kk = 0; kk < 4; ++kk) {
    // A fragment (16x32 f16): lane holds row m, K in [16hl,16hl+16)
    const _Float16* ap = xh + (size_t)(mt * 16 + m) * FIN + kk * 32 + hl * 16;
    v8h c1 = *(const v8h*)ap;
    v8h c2 = *(const v8h*)(ap + 8);
    v16h a;
#pragma unroll
    for (int i = 0; i < 8; ++i) { a[i] = c1[i]; a[i + 8] = c2[i]; }
    v16h b = wv[((size_t)(h * 4 + kk) * 4 + dt) * 32 + lane];
    c = __builtin_amdgcn_wmma_f32_16x16x32_f16(false, a, false, b, (short)0, c, false, false);
  }
  // f32 copy, node-major (for e_src/e_dst dots)
#pragma unroll
  for (int r = 0; r < 8; ++r) {
    int row = mt * 16 + r + 8 * hl;
    whf[((size_t)h * NN + row) * DH + dt * 16 + m] = c[r];
  }
  // f16 copy in B-fragment layout: frag=((h*128 + mt/2)*4 + dt),
  // element (lane_b = (mt&1)*16 + m, i = 8*hl + r) -> 8 contiguous halfs.
  v8h hv;
#pragma unroll
  for (int r = 0; r < 8; ++r) hv[r] = (_Float16)c[r];
  size_t fragb = ((size_t)h * 128 + (mt >> 1)) * 4 + dt;
  *(v8h*)(whh + fragb * 512 + ((mt & 1) * 16 + m) * 16 + 8 * hl) = hv;
}

// ---------------------------------------------------------------------------
// 3) e_src / e_dst = Wh . a_src / a_dst.  One wave per (head, node).
// ---------------------------------------------------------------------------
__global__ void k_edot(const float* __restrict__ whf, const float* __restrict__ as,
                       const float* __restrict__ ad, float* __restrict__ es,
                       float* __restrict__ ed) {
  int wave = (blockIdx.x * blockDim.x + threadIdx.x) >> 5;
  int lane = threadIdx.x & 31;
  if (wave >= NH * NN) return;
  int h = wave >> 12;                          // wave = h*NN + node
  const float* r = whf + (size_t)wave * DH;
  float v0 = r[lane], v1 = r[lane + 32];
  float ss = v0 * as[h * DH + lane] + v1 * as[h * DH + lane + 32];
  float sd = v0 * ad[h * DH + lane] + v1 * ad[h * DH + lane + 32];
  ss = warp_sum(ss);
  sd = warp_sum(sd);
  if (lane == 0) { es[wave] = ss; ed[wave] = sd; }
}

// ---------------------------------------------------------------------------
// 4) Fused multi-head flash-attention GAT layer.
//    Block = 96 threads = 3 waves (one per head).  adj tiles (16x32 f32) are
//    DMA'd by the TDM into double-buffered LDS (wave 0 issues, tensorcnt +
//    barrier synchronizes), overlapping the next tile's DMA with this tile's
//    online-softmax + WMMA aggregation  P(16x32) @ Wh(32x64).
// ---------------------------------------------------------------------------
__global__ void k_attn(const float* __restrict__ adj, const _Float16* __restrict__ whh,
                       const float* __restrict__ es, const float* __restrict__ ed,
                       float* __restrict__ hcat) {
  __shared__ float adjT[2][16 * 32];
  int rb = blockIdx.x;
  int w = threadIdx.x >> 5;                    // head
  int lane = threadIdx.x & 31;
  int m = lane & 15, hl = lane >> 4;
  float es_m = es[w * NN + rb * 16 + m];
  const float* edh = ed + (size_t)w * NN;
  const v16h* bv = (const v16h*)whh;
  const float* arow = adj + (size_t)(rb * 16) * NN;
  unsigned ldsA0 = (unsigned)(uintptr_t)&adjT[0][0];
  unsigned ldsA1 = (unsigned)(uintptr_t)&adjT[1][0];
  float mrow = NEGC, lrow = 0.f;
  v8f acc[4] = {};

  if (w == 0) tdm_load_adj_tile(arow, ldsA0);  // prologue: tile 0 -> buf 0

  for (int cb = 0; cb < NCH; ++cb) {
    int buf = cb & 1;
    if (w == 0) __builtin_amdgcn_s_wait_tensorcnt(0);
    __syncthreads();
    if (w == 0 && cb + 1 < NCH)                // DMA next tile into other buffer
      tdm_load_adj_tile(arow + (cb + 1) * 32, buf ? ldsA0 : ldsA1);
    if (cb + 1 < NCH)                          // prefetch next B fragments (L2 hint)
      __builtin_prefetch(&bv[(((size_t)w * 128 + cb + 1) * 4) * 32 + lane], 0, 0);

    float edl = edh[cb * 32 + lane];           // one column's e_dst per lane
    float ev[16];
    float tm = NEGC;
#pragma unroll
    for (int i = 0; i < 16; ++i) {
      int K = hl * 16 + i;   // A-fragment K map
      float e = es_m + __shfl(edl, K, 32);
      e = (e > 0.f) ? e : GALPHA * e;
      e = (adjT[buf][m * 32 + K] > 0.f) ? e : NEGC;
      ev[i] = e;
      tm = fmaxf(tm, e);
    }
    tm = fmaxf(tm, __shfl_xor(tm, 16, 32));    // combine K-halves of row m
    float newm = fmaxf(mrow, tm);
    float scale = __expf(mrow - newm);
    float psum = 0.f;
    v16h a;
#pragma unroll
    for (int i = 0; i < 16; ++i) {
      float p = __expf(ev[i] - newm);
      psum += p;
      a[i] = (_Float16)p;
    }
    psum += __shfl_xor(psum, 16, 32);
    lrow = lrow * scale + psum;
    mrow = newm;
    float sr[8];
#pragma unroll
    for (int r = 0; r < 8; ++r) sr[r] = __shfl(scale, r + 8 * hl, 32);  // row -> C layout
#pragma unroll
    for (int dtile = 0; dtile < 4; ++dtile) {
#pragma unroll
      for (int r = 0; r < 8; ++r) acc[dtile][r] *= sr[r];
      v16h b = bv[(((size_t)w * 128 + cb) * 4 + dtile) * 32 + lane];  // one 32-byte v16h load per lane
      acc[dtile] = __builtin_amdgcn_wmma_f32_16x16x32_f16(false, a, false, b, (short)0,
                                                          acc[dtile], false, false);
    }
  }
  float lr[8];
#pragma unroll
  for (int r = 0; r < 8; ++r) lr[r] = 1.f / __shfl(lrow, r + 8 * hl, 32);
#pragma unroll
  for (int dtile = 0; dtile < 4; ++dtile) {
#pragma unroll
    for (int r = 0; r < 8; ++r) {
      float v = acc[dtile][r] * lr[r];
      v = (v > 0.f) ? v : (__expf(v) - 1.f);   // elu
      int row = rb * 16 + r + 8 * hl;
      hcat[(size_t)row * (NH * DH) + w * DH + dtile * 16 + m] = v;
    }
  }
}

// ---------------------------------------------------------------------------
// 5) Wh2 = hcat @ W_out (192 -> 1), plus fs/fd = Wh2 * a_out.  One wave per node.
// ---------------------------------------------------------------------------
__global__ void k_outproj(const float* __restrict__ hcat, const float* __restrict__ Wout,
                          const float* __restrict__ aout, float* __restrict__ wh2,
                          float* __restrict__ fs, float* __restrict__ fd) {
  int wave = (blockIdx.x * blockDim.x + threadIdx.x) >> 5;
  int lane = threadIdx.x & 31;
  if (wave >= NN) return;
  const float* hr = hcat + (size_t)wave * (NH * DH);
  float s = 0.f;
#pragma unroll
  for (int j = 0; j < 6; ++j) s += hr[lane + 32 * j] * Wout[lane + 32 * j];
  s = warp_sum(s);
  if (lane == 0) {
    wh2[wave] = s;
    fs[wave]  = s * aout[0];
    fd[wave]  = s * aout[1];
  }
}

// ---------------------------------------------------------------------------
// 6) Output attention (D=1): scalar online softmax over one adj row per wave.
// ---------------------------------------------------------------------------
__global__ void k_outattn(const float* __restrict__ adj, const float* __restrict__ wh2,
                          const float* __restrict__ fs, const float* __restrict__ fd,
                          float* __restrict__ out) {
  int row = (blockIdx.x * blockDim.x + threadIdx.x) >> 5;
  int lane = threadIdx.x & 31;
  if (row >= NN) return;
  float fsi = fs[row];
  const float* arow = adj + (size_t)row * NN;
  float m = NEGC, l = 0.f, acc = 0.f;
  for (int j = lane; j < NN; j += 32) {
    float e = fsi + fd[j];
    e = (e > 0.f) ? e : GALPHA * e;
    e = (arow[j] > 0.f) ? e : NEGC;
    float nm = fmaxf(m, e);
    float sc = __expf(m - nm);
    float p  = __expf(e - nm);
    l = l * sc + p;
    acc = acc * sc + p * wh2[j];
    m = nm;
  }
#pragma unroll
  for (int off = 16; off; off >>= 1) {
    float m2 = __shfl_xor(m, off, 32);
    float l2 = __shfl_xor(l, off, 32);
    float a2 = __shfl_xor(acc, off, 32);
    float nm = fmaxf(m, m2);
    float s1 = __expf(m - nm), s2 = __expf(m2 - nm);
    l = l * s1 + l2 * s2;
    acc = acc * s1 + a2 * s2;
    m = nm;
  }
  if (lane == 0) {
    float o = acc / l;
    out[row] = (o > 0.f) ? o : (__expf(o) - 1.f);
  }
}

// ---------------------------------------------------------------------------
extern "C" void kernel_launch(void* const* d_in, const int* in_sizes, int n_in,
                              void* d_out, int out_size, void* d_ws, size_t ws_size,
                              hipStream_t stream) {
  const float* feat  = (const float*)d_in[0];   // (N, F)
  const float* adj   = (const float*)d_in[1];   // (N, N)
  const float* W     = (const float*)d_in[2];   // (H, F, D)
  const float* a_src = (const float*)d_in[3];   // (H, D)
  const float* a_dst = (const float*)d_in[4];   // (H, D)
  const float* W_out = (const float*)d_in[5];   // (H*D, 1)
  const float* a_out = (const float*)d_in[6];   // (2, 1)
  float* out = (float*)d_out;

  char* ws = (char*)d_ws;
  size_t off = 0;
  auto carve = [&](size_t bytes) -> void* {
    void* p = ws + off;
    off = (off + bytes + 255) & ~(size_t)255;
    return p;
  };
  _Float16* xh   = (_Float16*)carve((size_t)NN * FIN * sizeof(_Float16));       // 1 MB
  _Float16* wsw  = (_Float16*)carve((size_t)48 * 512 * sizeof(_Float16));       // 48 KB
  float*    whf  = (float*)   carve((size_t)NH * NN * DH * sizeof(float));      // 3 MB
  _Float16* whh  = (_Float16*)carve((size_t)NH * NN * DH * sizeof(_Float16));   // 1.5 MB
  float*    es   = (float*)   carve((size_t)NH * NN * sizeof(float));
  float*    ed   = (float*)   carve((size_t)NH * NN * sizeof(float));
  float*    hcat = (float*)   carve((size_t)NN * NH * DH * sizeof(float));      // 3 MB
  float*    wh2  = (float*)   carve((size_t)NN * sizeof(float));
  float*    fsb  = (float*)   carve((size_t)NN * sizeof(float));
  float*    fdb  = (float*)   carve((size_t)NN * sizeof(float));

  k_rownorm<<<NN / 4, 128, 0, stream>>>(feat, xh);
  k_prepw<<<12, 128, 0, stream>>>(W, wsw);
  k_proj<<<(256 * 4 * NH) / 8, 256, 0, stream>>>(xh, wsw, whf, whh);
  k_edot<<<(NH * NN) / 4, 128, 0, stream>>>(whf, a_src, a_dst, es, ed);
  k_attn<<<NN / 16, 96, 0, stream>>>(adj, whh, es, ed, hcat);
  k_outproj<<<NN / 4, 128, 0, stream>>>(hcat, W_out, a_out, wh2, fsb, fdb);
  k_outattn<<<NN / 4, 128, 0, stream>>>(adj, wh2, fsb, fdb, out);
}
